// MuralRestormer_36953898615387
// MI455X (gfx1250) — matrix operand order repaired and checked by reading, not compiled, not run on hardware
//
#include <hip/hip_runtime.h>
#include <hip/hip_bf16.h>

// ---------------------------------------------------------------------------
// MaxViT-style block: window attn + grid attn + channel gate, for MI455X
// (gfx1250). All matmuls use v_wmma_f32_16x16x32_bf16 (wave32 WMMA).
// Fragment traffic is organized as 16-byte vector loads so LDS/global paths
// lower to ds_load_b128 / global_load_b128 instead of scalar+pack.
// ---------------------------------------------------------------------------

typedef __bf16 bf16_t;
typedef __attribute__((ext_vector_type(16))) __bf16        v16bf;
typedef __attribute__((ext_vector_type(8)))  float         v8f;
typedef __attribute__((ext_vector_type(4)))  unsigned int  u32x4;

#define WSZ    8
#define NHEADS 4
#define CDIM   128
#define NTOK   64   // WSZ*WSZ
#define HDIM   32   // CDIM / NHEADS

// LDS leading dims (in halves / floats); all chosen so row strides are
// multiples of 16 bytes (136*2=272, 72*2=144, 68*4=272).
#define SXD 136     // 128-wide bf16 buffers (X, Q, K, O)
#define SVD 72      // V^T buffer: [128 channels][64 tokens + pad]
#define SSD 68      // 64-wide f32 logits
#define SPD 72      // 64-wide bf16 probs

#define SMEM_BYTES (4 * NTOK * SXD * 2 /*X,Q,K,O*/ + CDIM * SVD * 2 /*V^T*/ + \
                    NTOK * SSD * 4 /*S*/ + NTOK * SPD * 2 /*P*/ +             \
                    2 * 256 * 4 /*row max/sum reduction*/)

// ---- WMMA fragment helpers (wave32; A/B k = 16*(lane/16) + i, C/D per CDNA5 ISA 7.12.2) ---------

__device__ __forceinline__ v8f wmma_bf16(v16bf a, v16bf b, v8f c) {
  return __builtin_amdgcn_wmma_f32_16x16x32_bf16(
      /*neg_a=*/false, a, /*neg_b=*/false, b,
      /*c_mod=*/(short)0, c, /*reuse_a=*/false, /*reuse_b=*/false);
}

union FragBits {
  v16bf h;
  u32x4 q[2];
};

// Two 16-byte chunks -> one bf16 fragment (both pointers 16B aligned).
__device__ __forceinline__ v16bf ld_frag_2x16B(const bf16_t* p0, const bf16_t* p1) {
  FragBits f;
  f.q[0] = *(const u32x4*)p0;
  f.q[1] = *(const u32x4*)p1;
  return f.h;
}

// A: 16x32 bf16 tile, row-major source, leading dim ld (halves).
// Lane L<16 holds row L with K in {0..15}; lane L+16: K in {16..31}.
__device__ __forceinline__ v16bf load_frag_a(const bf16_t* base, int ld) {
  const int lane = threadIdx.x & 31;
  const int row  = lane & 15;
  const int khi  = (lane >> 4) << 4;    // 0 or 16
  const bf16_t* r = base + row * ld + khi;
  return ld_frag_2x16B(r, r + 8);
}

// B: 32x16 tile where logical B[k][n] = base[n*ld + k] (column-major source,
// i.e. K^T / V^T access); per lane 16 contiguous halves -> two 16B loads.
__device__ __forceinline__ v16bf load_frag_b_cm(const bf16_t* base, int ld) {
  const int lane = threadIdx.x & 31;
  const int col  = lane & 15;
  const int kb   = (lane >> 4) << 4;    // 0 or 16
  const bf16_t* r = base + col * ld + kb;
  return ld_frag_2x16B(r, r + 8);
}

// B fragment from pre-swizzled weights: tile stored [lane][16 halves].
__device__ __forceinline__ v16bf load_frag_b_swz(const bf16_t* swz, int tile) {
  const int lane = threadIdx.x & 31;
  const bf16_t* r = swz + (((tile << 5) + lane) << 4);  // (tile*32+lane)*16
  return ld_frag_2x16B(r, r + 8);
}

// C/D 16x16 f32: VGPR r -> (M = r + 8*(lane>=16), N = lane&15)
__device__ __forceinline__ void store_c_f32(float* base, int ld, v8f c) {
  const int lane = threadIdx.x & 31;
  const int col  = lane & 15;
  const int mo   = (lane >> 4) << 3;
#pragma unroll
  for (int r = 0; r < 8; ++r) base[(mo + r) * ld + col] = c[r];
}

__device__ __forceinline__ void store_c_bf16(bf16_t* base, int ld, v8f c, float scl) {
  const int lane = threadIdx.x & 31;
  const int col  = lane & 15;
  const int mo   = (lane >> 4) << 3;
#pragma unroll
  for (int r = 0; r < 8; ++r) base[(mo + r) * ld + col] = (bf16_t)(c[r] * scl);
}

// Transposed store: element (M,N) -> base[N*ld + M]; per lane 8 contiguous
// halves (16B, merges into a single ds_store_b128).
__device__ __forceinline__ void store_c_bf16_t(bf16_t* base, int ld, v8f c) {
  const int lane = threadIdx.x & 31;
  const int col  = lane & 15;
  const int mo   = (lane >> 4) << 3;
  bf16_t* p = base + col * ld + mo;
#pragma unroll
  for (int r = 0; r < 8; ++r) p[r] = (bf16_t)c[r];
}

// ---- fused per-window relative attention (window or grid partition) -------

template <bool GRID>
__global__ __launch_bounds__(256) void maxvit_attn_kernel(
    const float* __restrict__ xin, float* __restrict__ yout,
    const bf16_t* __restrict__ wqkv_swz,  // 96 tiles, fragment-swizzled
    const float* __restrict__ bqkv,
    const bf16_t* __restrict__ wproj_swz, // 32 tiles, fragment-swizzled
    const float* __restrict__ bproj,
    const float* __restrict__ bias,       // [NHEADS][64][64] precomputed rpb
    int B, int H, int W) {
  extern __shared__ char smem[];
  bf16_t* sX    = (bf16_t*)smem;              // [64][SXD]
  bf16_t* sQ    = sX + NTOK * SXD;            // Q (pre-scaled by 0.5)
  bf16_t* sK    = sQ + NTOK * SXD;
  bf16_t* sVt   = sK + NTOK * SXD;            // [128 ch][SVD] V transposed
  bf16_t* sO    = sVt + CDIM * SVD;           // attn output, heads concat
  float*  sS    = (float*)(sO + NTOK * SXD);  // [64][SSD] logits (one head)
  bf16_t* sP    = (bf16_t*)(sS + NTOK * SSD); // [64][SPD] probs  (one head)
  float*  sMaxR = (float*)(sP + NTOK * SPD);  // [4][64] segment maxes
  float*  sSumR = sMaxR + 256;                // [4][64] segment sums

  const int nbw  = W / WSZ;                 // blocks along W (=32)
  const int nwin = (H / WSZ) * nbw;         // windows per image (=1024)
  const int wid  = blockIdx.x;
  const int b    = wid / nwin;
  const int wrem = wid - b * nwin;
  const int bh   = wrem / nbw;
  const int bw   = wrem - bh * nbw;

  const int wave  = threadIdx.x >> 5;
  const int lane  = threadIdx.x & 31;
  const int col16 = lane & 15;
  const int mo    = (lane >> 4) << 3;

  auto gaddr = [&](int n, int c) -> long {
    const int i = n >> 3, j = n & 7;
    const int h = GRID ? (i * (H / WSZ) + bh) : (bh * WSZ + i);
    const int w = GRID ? (j * (W / WSZ) + bw) : (bw * WSZ + j);
    return (((long)(b * CDIM + c) * H) + h) * W + w;
  };

  // warm L2/L0 for the swizzled weights (global_prefetch_b8 path)
  __builtin_prefetch(wqkv_swz + ((long)threadIdx.x << 7), 0, 1);
  __builtin_prefetch(wproj_swz + ((long)threadIdx.x << 6), 0, 1);

  // ---- stage window into LDS as bf16 (token-major: lanes hit adjacent px) -
  for (int e = threadIdx.x; e < NTOK * CDIM; e += 256) {
    const int c = e >> 6, n = e & 63;
    sX[n * SXD + c] = (bf16_t)xin[gaddr(n, c)];
  }
  __syncthreads();

  // ---- QKV: [64,128] @ [128,384] + b ----------------------------------
  // Split into three uniform loops so each store target is static (no exec-
  // mask juggling). Weight column-tile indices: Q: nt, K: nt+8, V: nt+16.
#pragma unroll
  for (int t = wave; t < 32; t += 8) {  // Q
    const int mt = t >> 3, nt = t & 7;
    const float bq = bqkv[(nt << 4) + col16];
    v8f acc;
#pragma unroll
    for (int r = 0; r < 8; ++r) acc[r] = bq;
#pragma unroll
    for (int kk = 0; kk < 4; ++kk) {
      v16bf a  = load_frag_a(sX + (mt << 4) * SXD + (kk << 5), SXD);
      v16bf bf = load_frag_b_swz(wqkv_swz, kk * 24 + nt);
      acc = wmma_bf16(a, bf, acc);
    }
    store_c_bf16(sQ + (mt << 4) * SXD + (nt << 4), SXD, acc, 0.5f);  // scale folded
  }
#pragma unroll
  for (int t = wave; t < 32; t += 8) {  // K
    const int mt = t >> 3, nt = t & 7;
    const float bq = bqkv[CDIM + (nt << 4) + col16];
    v8f acc;
#pragma unroll
    for (int r = 0; r < 8; ++r) acc[r] = bq;
#pragma unroll
    for (int kk = 0; kk < 4; ++kk) {
      v16bf a  = load_frag_a(sX + (mt << 4) * SXD + (kk << 5), SXD);
      v16bf bf = load_frag_b_swz(wqkv_swz, kk * 24 + 8 + nt);
      acc = wmma_bf16(a, bf, acc);
    }
    store_c_bf16(sK + (mt << 4) * SXD + (nt << 4), SXD, acc, 1.0f);
  }
#pragma unroll
  for (int t = wave; t < 32; t += 8) {  // V (stored transposed [ch][token])
    const int mt = t >> 3, nt = t & 7;
    const float bq = bqkv[2 * CDIM + (nt << 4) + col16];
    v8f acc;
#pragma unroll
    for (int r = 0; r < 8; ++r) acc[r] = bq;
#pragma unroll
    for (int kk = 0; kk < 4; ++kk) {
      v16bf a  = load_frag_a(sX + (mt << 4) * SXD + (kk << 5), SXD);
      v16bf bf = load_frag_b_swz(wqkv_swz, kk * 24 + 16 + nt);
      acc = wmma_bf16(a, bf, acc);
    }
    store_c_bf16_t(sVt + (nt << 4) * SVD + (mt << 4), SVD, acc);
  }
  __syncthreads();

  // ---- per-head attention -------------------------------------------------
  const int srow = threadIdx.x & 63;   // softmax row
  const int sseg = threadIdx.x >> 6;   // 16-col segment (0..3)
  for (int h = 0; h < NHEADS; ++h) {
    // logits S = Qh @ Kh^T : 16 tiles of 16x16, K=32 (one WMMA each)
#pragma unroll
    for (int tt = 0; tt < 2; ++tt) {
      const int t = wave + (tt << 3);
      const int mt = t >> 2, nt = t & 3;
      v16bf a  = load_frag_a(sQ + (mt << 4) * SXD + (h << 5), SXD);
      v16bf bf = load_frag_b_cm(sK + (nt << 4) * SXD + (h << 5), SXD);
      v8f acc;
#pragma unroll
      for (int r = 0; r < 8; ++r) acc[r] = 0.0f;
      acc = wmma_bf16(a, bf, acc);
      store_c_f32(sS + (mt << 4) * SSD + (nt << 4), SSD, acc);
    }
    __syncthreads();

    // softmax, all 256 threads: 4 threads per row x 16 columns
    {
      const float* brow = bias + (((h << 6) + srow) << 6) + (sseg << 4);
      float* swrow = sS + srow * SSD + (sseg << 4);
      float mx = -3.0e30f;
#pragma unroll
      for (int m = 0; m < 16; ++m) mx = fmaxf(mx, swrow[m] + brow[m]);
      sMaxR[(sseg << 6) + srow] = mx;
      __syncthreads();
      mx = fmaxf(fmaxf(sMaxR[srow], sMaxR[64 + srow]),
                 fmaxf(sMaxR[128 + srow], sMaxR[192 + srow]));
      float sum = 0.0f;
#pragma unroll
      for (int m = 0; m < 16; ++m) {
        const float e = __expf(swrow[m] + brow[m] - mx);
        swrow[m] = e;  // thread-exclusive 16-element segment
        sum += e;
      }
      sSumR[(sseg << 6) + srow] = sum;
      __syncthreads();
      const float inv = 1.0f / (sSumR[srow] + sSumR[64 + srow] +
                                sSumR[128 + srow] + sSumR[192 + srow]);
      bf16_t* prow = sP + srow * SPD + (sseg << 4);
#pragma unroll
      for (int m = 0; m < 16; ++m) prow[m] = (bf16_t)(swrow[m] * inv);
    }
    __syncthreads();

    // Oh = P @ Vh : 8 tiles (one per wave), K=64 (2 WMMAs); V^T gives
    // contiguous per-lane B fragments.
    {
      const int mt = wave >> 1, nt = wave & 1;
      v8f acc;
#pragma unroll
      for (int r = 0; r < 8; ++r) acc[r] = 0.0f;
#pragma unroll
      for (int kk = 0; kk < 2; ++kk) {
        v16bf a  = load_frag_a(sP + (mt << 4) * SPD + (kk << 5), SPD);
        v16bf bf = load_frag_b_cm(sVt + ((h << 5) + (nt << 4)) * SVD + (kk << 5), SVD);
        acc = wmma_bf16(a, bf, acc);
      }
      store_c_bf16(sO + (mt << 4) * SXD + (h << 5) + (nt << 4), SXD, acc, 1.0f);
    }
    __syncthreads();
  }

  // ---- output proj + bias + residual -> global ----------------------------
  for (int t = wave; t < 32; t += 8) {
    const int mt = t >> 3, nt = t & 7;
    const int ncol = nt << 4;
    const float bp = bproj[ncol + col16];
    v8f acc;
#pragma unroll
    for (int r = 0; r < 8; ++r) acc[r] = bp;
#pragma unroll
    for (int kk = 0; kk < 4; ++kk) {
      v16bf a  = load_frag_a(sO + (mt << 4) * SXD + (kk << 5), SXD);
      v16bf bf = load_frag_b_swz(wproj_swz, kk * 8 + nt);
      acc = wmma_bf16(a, bf, acc);
    }
    // residual: each element is read then written by exactly one lane, so the
    // grid pass can run in place (yout == xin) safely.
#pragma unroll
    for (int r = 0; r < 8; ++r) {
      const int n = (mt << 4) + mo + r;
      const int c = ncol + col16;
      const long g = gaddr(n, c);
      yout[g] = xin[g] + acc[r];
    }
  }
}

// ---- weight swizzle: row-major [K][N] f32 -> per-tile fragment layout -----
// dst[((kt*ntn + nt)*32 + lane)*16 + i] = B[kt*32 + kb(lane) + i][nt*16 + col(lane)]

__global__ void swz_b_kernel(const float* __restrict__ src, bf16_t* __restrict__ dst,
                             int N, int ntn, int total) {
  const int tid = blockIdx.x * blockDim.x + threadIdx.x;
  if (tid >= total) return;
  const int i    = tid & 15;
  const int lane = (tid >> 4) & 31;
  const int tile = tid >> 9;
  const int kt = tile / ntn, nt = tile - kt * ntn;
  const int col = lane & 15;
  const int kb  = (lane >> 4) << 4;
  dst[tid] = (bf16_t)src[(kt * 32 + kb + i) * N + nt * 16 + col];
}

// ---- relative-position bias expansion: rpb[225][H] -> bias[H][64][64] -----

__global__ void bias_expand_kernel(const float* __restrict__ rpb,
                                   float* __restrict__ bias) {
  const int tid = blockIdx.x * blockDim.x + threadIdx.x;  // < 4*64*64
  if (tid >= NHEADS * NTOK * NTOK) return;
  const int m = tid & 63;
  const int n = (tid >> 6) & 63;
  const int h = tid >> 12;
  const int i1 = n >> 3, j1 = n & 7;
  const int i2 = m >> 3, j2 = m & 7;
  const int ridx = (i1 - i2 + WSZ - 1) * (2 * WSZ - 1) + (j1 - j2 + WSZ - 1);
  bias[tid] = rpb[ridx * NHEADS + h];
}

// ---- channel-gate helpers -------------------------------------------------

__global__ __launch_bounds__(256) void stats_kernel(
    const float* __restrict__ x, float* __restrict__ avg,
    float* __restrict__ mx, int HW) {
  const int bc = blockIdx.x;  // b*C + c
  const float* p = x + (long)bc * HW;
  float s = 0.0f, m = -3.0e30f;
  for (int i = threadIdx.x; i < HW; i += 256) {
    const float v = p[i];
    s += v;
    m = fmaxf(m, v);
  }
  __shared__ float ss[256];
  __shared__ float sm[256];
  ss[threadIdx.x] = s;
  sm[threadIdx.x] = m;
  __syncthreads();
  for (int o = 128; o > 0; o >>= 1) {
    if (threadIdx.x < o) {
      ss[threadIdx.x] += ss[threadIdx.x + o];
      sm[threadIdx.x] = fmaxf(sm[threadIdx.x], sm[threadIdx.x + o]);
    }
    __syncthreads();
  }
  if (threadIdx.x == 0) {
    avg[bc] = ss[0] / (float)HW;
    mx[bc]  = sm[0];
  }
}

__global__ __launch_bounds__(128) void gate_kernel(
    const float* __restrict__ avg, const float* __restrict__ mx,
    const float* __restrict__ w1, const float* __restrict__ w2,
    float* __restrict__ gate, int B) {
  __shared__ float hsum[4][CDIM];  // relu(avg@W1) + relu(max@W1)
  const int t = threadIdx.x;
  for (int b = 0; b < B; ++b) {
    float ha = 0.0f, hm = 0.0f;
    for (int c = 0; c < CDIM; ++c) {
      const float w = w1[c * CDIM + t];
      ha += avg[b * CDIM + c] * w;
      hm += mx[b * CDIM + c] * w;
    }
    hsum[b][t] = fmaxf(ha, 0.0f) + fmaxf(hm, 0.0f);
  }
  __syncthreads();
  for (int b = 0; b < B; ++b) {
    float g = 0.0f;
    for (int k = 0; k < CDIM; ++k) g += hsum[b][k] * w2[k * CDIM + t];
    gate[b * CDIM + t] = 1.0f / (1.0f + __expf(-g));
  }
}

__global__ void scale_kernel(float* __restrict__ x,
                             const float* __restrict__ gate,
                             long HW, long total) {
  const long i = (long)blockIdx.x * blockDim.x + threadIdx.x;
  if (i < total) x[i] *= gate[i / HW];  // i/HW == b*C + c for [B,C,H,W]
}

// ---------------------------------------------------------------------------

extern "C" void kernel_launch(void* const* d_in, const int* in_sizes, int n_in,
                              void* d_out, int out_size, void* d_ws, size_t ws_size,
                              hipStream_t stream) {
  (void)in_sizes; (void)n_in; (void)out_size; (void)ws_size;
  const float* x     = (const float*)d_in[0];
  const float* wqkv  = (const float*)d_in[1];
  const float* bqkv  = (const float*)d_in[2];
  const float* wproj = (const float*)d_in[3];
  const float* bproj = (const float*)d_in[4];
  const float* rpb   = (const float*)d_in[5];
  const float* cfw1  = (const float*)d_in[6];
  const float* cfw2  = (const float*)d_in[7];
  float* y = (float*)d_out;

  const int B = 4, C = CDIM, H = 256, W = 256;

  // workspace carve (rebuilt every call; graph-capture safe)
  char* ws = (char*)d_ws;
  bf16_t* wqkv_swz  = (bf16_t*)ws;                       // 128*384 halves
  bf16_t* wproj_swz = (bf16_t*)(ws + C * 3 * C * 2);     // 128*128 halves
  float*  avg       = (float*)(ws + C * 3 * C * 2 + C * C * 2);
  float*  mxv       = avg + B * C;
  float*  gate      = mxv + B * C;
  float*  bias      = gate + B * C;                      // [4][64][64] f32

  const int nq = C * 3 * C;  // 49152 fragment elements (96 tiles)
  const int np = C * C;      // 16384 fragment elements (32 tiles)
  swz_b_kernel<<<(nq + 255) / 256, 256, 0, stream>>>(wqkv, wqkv_swz, 3 * C, 24, nq);
  swz_b_kernel<<<(np + 255) / 256, 256, 0, stream>>>(wproj, wproj_swz, C, 8, np);
  const int nb = NHEADS * NTOK * NTOK;  // 16384
  bias_expand_kernel<<<(nb + 255) / 256, 256, 0, stream>>>(rpb, bias);

  const int nwin = B * (H / WSZ) * (W / WSZ);  // 4096 workgroups
  // pass 1: window attention, x -> y
  maxvit_attn_kernel<false><<<nwin, 256, SMEM_BYTES, stream>>>(
      x, y, wqkv_swz, bqkv, wproj_swz, bproj, bias, B, H, W);
  // pass 2: grid attention, in place on y (disjoint per-window pixel sets)
  maxvit_attn_kernel<true><<<nwin, 256, SMEM_BYTES, stream>>>(
      y, y, wqkv_swz, bqkv, wproj_swz, bproj, bias, B, H, W);

  // channel gate
  stats_kernel<<<B * C, 256, 0, stream>>>(y, avg, mxv, H * W);
  gate_kernel<<<1, 128, 0, stream>>>(avg, mxv, cfw1, cfw2, gate, B);
  const long total = (long)B * C * H * W;
  scale_kernel<<<(int)((total + 255) / 256), 256, 0, stream>>>(y, gate, (long)H * W, total);
}
